// GraphAttentionLayerV2_75797582840269
// MI455X (gfx1250) — matrix operand order repaired and checked by reading, not compiled, not run on hardware
//
#include <hip/hip_runtime.h>
#include <hip/hip_bf16.h>
#include <math.h>

#define SLOPE   0.2f
#define N_NODES 10000
#define N_EDGES 320000
#define D_NODE  256
#define D_EDGE  128
#define D_HID   256

typedef __attribute__((ext_vector_type(16))) __bf16 v16bf;
typedef __attribute__((ext_vector_type(8)))  __bf16 v8bf;
typedef __attribute__((ext_vector_type(8)))  float  v8f;

// Hardware f32->bf16 (v_cvt_pk_bf16_f32), RNE.
__device__ __forceinline__ __bf16 f2bf(float f) { return (__bf16)f; }
__device__ __forceinline__ float lrelu(float x) { return x > 0.f ? x : SLOPE * x; }

// 16x32 bf16 A-fragment from row-major fp32 (contiguous-K order, matched by B), wave32.
// lane L: M = L&15 (caller passes that row); e: K=kk+16*l16+e
__device__ __forceinline__ v16bf load_a_frag(const float* __restrict__ A, int row,
                                             int kk, int ldk, int l16) {
  const float* p0 = A + (size_t)row * ldk + kk + l16 * 16;
  float4 x0 = *(const float4*)(p0);
  float4 x1 = *(const float4*)(p0 + 4);
  float4 y0 = *(const float4*)(p0 + 8);
  float4 y1 = *(const float4*)(p0 + 12);
  v16bf a;
  a[0]=f2bf(x0.x);  a[1]=f2bf(x0.y);  a[2]=f2bf(x0.z);  a[3]=f2bf(x0.w);
  a[4]=f2bf(x1.x);  a[5]=f2bf(x1.y);  a[6]=f2bf(x1.z);  a[7]=f2bf(x1.w);
  a[8]=f2bf(y0.x);  a[9]=f2bf(y0.y);  a[10]=f2bf(y0.z); a[11]=f2bf(y0.w);
  a[12]=f2bf(y1.x); a[13]=f2bf(y1.y); a[14]=f2bf(y1.z); a[15]=f2bf(y1.w);
  return a;
}

// B-fragment from LDS in CHUNK-MAJOR layout: Bt[c][n][16], c = k/16 (32B K-chunks).
// lane L: N = L&15 -> consecutive lanes read consecutive 32B chunks: bank-conflict-free.
__device__ __forceinline__ v16bf load_b_frag(const __bf16* Bt, int n, int kk,
                                             int ncols, int l16) {
  return *(const v16bf*)(Bt + (size_t)(((kk >> 4) + l16) * ncols + n) * 16);
}

// Cooperative weight staging into chunk-major bf16 LDS.
// W is [K][D_HID] fp32 row-major; we stage columns [n0, n0+ncols), all K.
// Each work-item: 8 K-strided coalesced global loads -> 4 cvt_pk -> one ds_store_b128.
__device__ __forceinline__ void stage_weights(const float* __restrict__ W, __bf16* Bt,
                                              int n0, int ncols, int K, int t) {
  const int items = (K >> 3) * ncols;          // one item = 8 k-values of one column
  for (int idx = t; idx < items; idx += 256) {
    int n  = idx % ncols;
    int k0 = (idx / ncols) << 3;
    v8bf d;
#pragma unroll
    for (int i = 0; i < 8; ++i) d[i] = f2bf(W[(size_t)(k0 + i) * D_HID + n0 + n]);
    // byte offset = (k0/16)*ncols*32 + n*32 + (k0&15)*2  -> 16B aligned
    *(v8bf*)(Bt + (size_t)((k0 >> 4) * ncols + n) * 16 + (k0 & 15)) = d;
  }
}

// ---- Kernel 1: h = lrelu(X @ W + b)   [10000,256] x [256,256] -------------
// Block: 8 waves x 16 rows = 128 rows, 128 cols (8 n-tiles per wave). grid.y = 2.
__global__ __launch_bounds__(256) void k_node_proj(const float* __restrict__ X,
                                                   const float* __restrict__ W,
                                                   const float* __restrict__ b,
                                                   float* __restrict__ H) {
  __shared__ __align__(32) __bf16 Bt[128 * D_NODE];   // 64 KB, chunk-major
  const int t = threadIdx.x;
  const int n0 = blockIdx.y * 128;
  stage_weights(W, Bt, n0, 128, D_NODE, t);
  __syncthreads();
  const int lane = t & 31, wave = t >> 5, l16 = lane >> 4, nl = lane & 15;
  const int rowbase = blockIdx.x * 128 + wave * 16;
  int row = rowbase + nl;
  int rowc = row < N_NODES ? row : N_NODES - 1;
  v8f acc[8] = {};
#pragma unroll 2
  for (int kk = 0; kk < D_NODE; kk += 32) {
    v16bf a = load_a_frag(X, rowc, kk, D_NODE, l16);
#pragma unroll
    for (int j = 0; j < 8; ++j) {
      v16bf bf = load_b_frag(Bt, j * 16 + nl, kk, 128, l16);
      acc[j] = __builtin_amdgcn_wmma_f32_16x16x32_bf16(false, a, false, bf,
                                                       (short)0, acc[j], false, false);
    }
  }
  if (rowbase + 16 <= N_NODES) {               // fast path, no per-element guard
#pragma unroll
    for (int j = 0; j < 8; ++j) {
      int n = n0 + j * 16 + nl;
      float bias = b[n];
#pragma unroll
      for (int r = 0; r < 8; ++r)
        H[(size_t)(rowbase + r + 8 * l16) * D_HID + n] = lrelu(acc[j][r] + bias);
    }
  } else {
#pragma unroll
    for (int j = 0; j < 8; ++j) {
      int n = n0 + j * 16 + nl;
      float bias = b[n];
#pragma unroll
      for (int r = 0; r < 8; ++r) {
        int m = rowbase + r + 8 * l16;
        if (m < N_NODES) H[(size_t)m * D_HID + n] = lrelu(acc[j][r] + bias);
      }
    }
  }
}

// ---- Kernel 2: s3[e] = lrelu(EF @ We + be) . a3   (fused, edge_h never stored)
// Block: stages weights ONCE, then runs 4 row-tiles of 128 edges (8 waves x 16 rows).
// Each wave owns all 256 cols (16 n-tiles) -> a3-dot reduces fully in-wave.
#define EDGE_TILES 4
__global__ __launch_bounds__(256) void k_edge_proj_dot(const float* __restrict__ EF,
                                                       const float* __restrict__ W,
                                                       const float* __restrict__ b,
                                                       const float* __restrict__ a3,
                                                       float* __restrict__ S3) {
  __shared__ __align__(32) __bf16 Bt[D_HID * D_EDGE];  // 64 KB, chunk-major
  const int t = threadIdx.x;
  stage_weights(W, Bt, 0, D_HID, D_EDGE, t);
  __syncthreads();
  const int lane = t & 31, wave = t >> 5, l16 = lane >> 4, nl = lane & 15;
  for (int it = 0; it < EDGE_TILES; ++it) {    // E = 128 * 4 * 625, no tail
    const int rowbase = (blockIdx.x * EDGE_TILES + it) * 128 + wave * 16;
    const int row = rowbase + nl;
    v8f acc[16] = {};
    for (int kk = 0; kk < D_EDGE; kk += 32) {
      v16bf a = load_a_frag(EF, row, kk, D_EDGE, l16);
#pragma unroll
      for (int j = 0; j < 16; ++j) {
        v16bf bf = load_b_frag(Bt, j * 16 + nl, kk, D_HID, l16);
        acc[j] = __builtin_amdgcn_wmma_f32_16x16x32_bf16(false, a, false, bf,
                                                         (short)0, acc[j], false, false);
      }
    }
    float pr[8] = {};
#pragma unroll
    for (int j = 0; j < 16; ++j) {
      int n = j * 16 + nl;
      float bias = b[n], w3 = a3[n];
#pragma unroll
      for (int r = 0; r < 8; ++r) pr[r] += lrelu(acc[j][r] + bias) * w3;
    }
#pragma unroll
    for (int r = 0; r < 8; ++r) {              // reduce over the 16 N-lanes
      float v = pr[r];
      v += __shfl_xor(v, 1); v += __shfl_xor(v, 2);
      v += __shfl_xor(v, 4); v += __shfl_xor(v, 8);
      if (nl == 0) S3[rowbase + 8 * l16 + r] = v;
    }
  }
}

// ---- Kernel 3: per-node scores s1 = h.a1, s2 = h.a2 (one wave / node) -----
__global__ __launch_bounds__(256) void k_node_scores(const float* __restrict__ H,
                                                     const float* __restrict__ a1,
                                                     const float* __restrict__ a2,
                                                     float* __restrict__ S1,
                                                     float* __restrict__ S2) {
  int w = (blockIdx.x * 256 + threadIdx.x) >> 5;
  int lane = threadIdx.x & 31;
  if (w >= N_NODES) return;
  const float* hp = H + (size_t)w * D_HID;
  float v1 = 0.f, v2 = 0.f;
  for (int k = lane; k < D_HID; k += 32) { float h = hp[k]; v1 += h * a1[k]; v2 += h * a2[k]; }
  for (int m = 16; m; m >>= 1) { v1 += __shfl_xor(v1, m); v2 += __shfl_xor(v2, m); }
  if (lane == 0) { S1[w] = v1; S2[w] = v2; }
}

// ---- Init: segmax=-inf, segsum=0, out=0 -----------------------------------
__global__ void k_init(float* segmax, float* segsum, float* out) {
  int i = blockIdx.x * blockDim.x + threadIdx.x;
  if (i < N_NODES) { segmax[i] = -__builtin_huge_valf(); segsum[i] = 0.f; }
  if (i < N_NODES * D_HID) out[i] = 0.f;
}

__device__ __forceinline__ void atomicMaxF(float* addr, float val) {
  if (val >= 0.f) atomicMax((int*)addr, __float_as_int(val));
  else            atomicMin((unsigned int*)addr, __float_as_uint(val));
}

// ---- Kernel 4: logits + segment max ---------------------------------------
__global__ void k_logit_max(const int* __restrict__ EI, const float* __restrict__ S1,
                            const float* __restrict__ S2, const float* __restrict__ S3,
                            const float* __restrict__ AB, float* __restrict__ EL,
                            float* __restrict__ segmax) {
  int e = blockIdx.x * blockDim.x + threadIdx.x;
  if (e >= N_EDGES) return;
  int tgt = EI[e], nbr = EI[N_EDGES + e];
  float v = S1[tgt] + S2[nbr] + S3[e] + AB[0];
  EL[e] = v;
  atomicMaxF(&segmax[tgt], v);
}

// ---- Kernel 5: exp + segment sum ------------------------------------------
__global__ void k_exp_sum(const int* __restrict__ EI, float* __restrict__ EL,
                          const float* __restrict__ segmax, float* __restrict__ segsum) {
  int e = blockIdx.x * blockDim.x + threadIdx.x;
  if (e >= N_EDGES) return;
  int tgt = EI[e];
  float ex = __expf(EL[e] - segmax[tgt]);
  EL[e] = ex;
  atomicAdd(&segsum[tgt], ex);
}

// ---- Kernel 6: scatter out[tgt] += alpha * h[nbr] (one wave / edge) -------
__global__ __launch_bounds__(256) void k_scatter(const int* __restrict__ EI,
                                                 const float* __restrict__ EL,
                                                 const float* __restrict__ segsum,
                                                 const float* __restrict__ H,
                                                 float* __restrict__ out) {
  int e = (blockIdx.x * 256 + threadIdx.x) >> 5;
  int lane = threadIdx.x & 31;
  if (e >= N_EDGES) return;
  int tgt = EI[e], nbr = EI[N_EDGES + e];
  float alpha = EL[e] / segsum[tgt];
  const float* hj = H + (size_t)nbr * D_HID;
  float* op = out + (size_t)tgt * D_HID;
#pragma unroll
  for (int k = 0; k < D_HID; k += 32)
    atomicAdd(&op[k + lane], alpha * hj[k + lane]);
}

// ---- Kernel 7: final lrelu -------------------------------------------------
__global__ void k_final(float* __restrict__ out) {
  int i = blockIdx.x * blockDim.x + threadIdx.x;
  if (i < N_NODES * D_HID) out[i] = lrelu(out[i]);
}

extern "C" void kernel_launch(void* const* d_in, const int* in_sizes, int n_in,
                              void* d_out, int out_size, void* d_ws, size_t ws_size,
                              hipStream_t stream) {
  const float* nodeF = (const float*)d_in[0];
  const float* edgeF = (const float*)d_in[1];
  const int*   ei    = (const int*)d_in[2];
  const float* wW    = (const float*)d_in[3];
  const float* wb    = (const float*)d_in[4];
  const float* ewW   = (const float*)d_in[5];
  const float* ewb   = (const float*)d_in[6];
  const float* aW    = (const float*)d_in[7];   // [768]: a1 | a2 | a3
  const float* ab    = (const float*)d_in[8];
  float* out = (float*)d_out;

  float* h      = (float*)d_ws;                         // N*H
  float* s1     = h + (size_t)N_NODES * D_HID;          // N
  float* s2     = s1 + N_NODES;                         // N
  float* s3     = s2 + N_NODES;                         // E
  float* el     = s3 + N_EDGES;                         // E
  float* segmax = el + N_EDGES;                         // N
  float* segsum = segmax + N_NODES;                     // N

  dim3 g1((N_NODES + 127) / 128, D_HID / 128);
  k_node_proj<<<g1, 256, 0, stream>>>(nodeF, wW, wb, h);
  k_edge_proj_dot<<<N_EDGES / (128 * EDGE_TILES), 256, 0, stream>>>(edgeF, ewW, ewb,
                                                                    aW + 2 * D_HID, s3);
  k_node_scores<<<(N_NODES * 32 + 255) / 256, 256, 0, stream>>>(h, aW, aW + D_HID, s1, s2);
  k_init<<<(N_NODES * D_HID + 255) / 256, 256, 0, stream>>>(segmax, segsum, out);
  k_logit_max<<<(N_EDGES + 255) / 256, 256, 0, stream>>>(ei, s1, s2, s3, ab, el, segmax);
  k_exp_sum<<<(N_EDGES + 255) / 256, 256, 0, stream>>>(ei, el, segmax, segsum);
  k_scatter<<<(N_EDGES * 32 + 255) / 256, 256, 0, stream>>>(ei, el, segsum, h, out);
  k_final<<<(N_NODES * D_HID + 255) / 256, 256, 0, stream>>>(out);
}
